// HammingDistanceAttention_6932077216007
// MI455X (gfx1250) — compile-verified, matrix operand order checked by reading, not run on hardware
//
#include <hip/hip_runtime.h>

typedef __attribute__((ext_vector_type(16))) _Float16 v16h;
typedef __attribute__((ext_vector_type(8)))  _Float16 v8h;
typedef __attribute__((ext_vector_type(4)))  _Float16 v4h;
typedef __attribute__((ext_vector_type(8)))  float    v8f;

#define Bsz 4
#define Hn  8
#define Nn  2048
#define Dd  64
#define NT  128                 // key tiles of 16
#define HEAD_ELEMS (Nn*Dd)      // 131072

// exp(0.5 + c/128) = 2^( c * S2 + B2 )
#define LOG2E 1.4426950408889634f
#define S2    (LOG2E / 128.0f)
#define B2    (0.5f * LOG2E)

__device__ __forceinline__ v8f wmma_f16(v16h a, v16h b, v8f c) {
    return __builtin_amdgcn_wmma_f32_16x16x32_f16(false, a, false, b, (short)0, c,
                                                  false, false);
}

// Raw v_exp_f32: safe here because our exponent arg is always > -13,
// far from the -126 subnormal region the libm expansion guards against.
__device__ __forceinline__ float fast_exp2(float x) {
    return __builtin_amdgcn_exp2f(x);
}

// xor-add reduce across 16-lane halves of a wave32 (masks 1,2,4,8)
__device__ __forceinline__ float half16_sum(float v) {
    int i;
    i = __builtin_amdgcn_ds_swizzle(__float_as_int(v), 0x041F); v += __int_as_float(i);
    i = __builtin_amdgcn_ds_swizzle(__float_as_int(v), 0x081F); v += __int_as_float(i);
    i = __builtin_amdgcn_ds_swizzle(__float_as_int(v), 0x101F); v += __int_as_float(i);
    i = __builtin_amdgcn_ds_swizzle(__float_as_int(v), 0x201F); v += __int_as_float(i);
    return v;
}

// 16x32 (MxK) 16-bit A/B fragment from f32 row-major row pointer, y = 2x-1.
// ISA 7.12.2: lane (l<16) holds K in {8H..8H+7} U {16+8H..16+8H+7}, H=lane>>4.
__device__ __forceinline__ v16h load_frag_f32_pm1(const float* p, int H) {
    float f[16];
    *(float4*)(f + 0)  = *(const float4*)(p + 8 * H);
    *(float4*)(f + 4)  = *(const float4*)(p + 8 * H + 4);
    *(float4*)(f + 8)  = *(const float4*)(p + 16 + 8 * H);
    *(float4*)(f + 12) = *(const float4*)(p + 16 + 8 * H + 4);
    v16h r;
#pragma unroll
    for (int j = 0; j < 16; ++j) r[j] = (_Float16)(2.0f * f[j] - 1.0f);
    return r;
}

// Same fragment layout from a pre-converted f16 row-major row pointer.
__device__ __forceinline__ v16h load_frag_f16(const _Float16* p, int H) {
    v8h lo = *(const v8h*)(p + 8 * H);
    v8h hi = *(const v8h*)(p + 16 + 8 * H);
    v16h r;
#pragma unroll
    for (int j = 0; j < 8; ++j) { r[j] = lo[j]; r[j + 8] = hi[j]; }
    return r;
}

// ---- prep 1: Kh[bh][n][k] = (f16)(2*K - 1), coalesced float4 -> half4 ----
__global__ __launch_bounds__(256) void kprep_kernel(const float* __restrict__ k,
                                                    _Float16* __restrict__ kh) {
    size_t i = ((size_t)blockIdx.x * 256 + threadIdx.x) * 4;
    float4 f = *(const float4*)(k + i);
    v4h h;
    h[0] = (_Float16)(2.0f * f.x - 1.0f);
    h[1] = (_Float16)(2.0f * f.y - 1.0f);
    h[2] = (_Float16)(2.0f * f.z - 1.0f);
    h[3] = (_Float16)(2.0f * f.w - 1.0f);
    *(v4h*)(kh + i) = h;
}

// ---- prep 2: V [bh][n][d] f32  ->  Vt [bh][d][n] f16 (tiled transpose) ----
__global__ __launch_bounds__(256) void vtrans_kernel(const float* __restrict__ v,
                                                     _Float16* __restrict__ vt) {
    __shared__ float t[64][65];
    int bh = blockIdx.x >> 5;            // 0..31
    int n0 = (blockIdx.x & 31) * 64;     // n tile
    int c  = threadIdx.x & 63;
    int rq = threadIdx.x >> 6;           // 0..3
    const float* src = v + ((size_t)bh * Nn + n0) * Dd;
#pragma unroll
    for (int i = 0; i < 16; ++i) {
        int r = rq + i * 4;
        t[r][c] = src[(size_t)r * Dd + c];
    }
    __syncthreads();
    _Float16* dst = vt + (size_t)bh * Dd * Nn + n0;
#pragma unroll
    for (int i = 0; i < 16; ++i) {
        int d = rq + i * 4;
        dst[(size_t)d * Nn + c] = (_Float16)t[c][d];
    }
}

// ---- main kernel: one wave = one (bh, 16-row query tile) ----
__global__ __launch_bounds__(256) void hamm_attn_kernel(
        const float* __restrict__ q, const _Float16* __restrict__ kh,
        const _Float16* __restrict__ vt, float* __restrict__ out,
        float* __restrict__ score) {
    // per-wave P staging: 16 rows x 40 halves (80B stride: 16B aligned,
    // bank-conflict-free for ds_load_b128)
    __shared__ _Float16 lds[8 * 16 * 40];

    const int lane = threadIdx.x & 31;
    const int wave = threadIdx.x >> 5;
    const int task = blockIdx.x * 8 + wave;     // 0..4095
    const int bh   = task >> 7;                 // 0..31
    const int qt   = task & 127;                // query tile
    const int nlo  = lane & 15;
    const int H    = lane >> 4;

    const float*    qrow  = q  + (size_t)bh * HEAD_ELEMS + (size_t)(qt * 16 + nlo) * Dd;
    const _Float16* kbase = kh + (size_t)bh * HEAD_ELEMS;
    const _Float16* vbase = vt + (size_t)bh * HEAD_ELEMS;
    float* scb = score + ((size_t)bh * Nn + qt * 16) * Nn;
    float* ob  = out   + ((size_t)bh * Nn + qt * 16) * Dd;

    const v16h a0 = load_frag_f32_pm1(qrow, H);        // K = 0..31
    const v16h a1 = load_frag_f32_pm1(qrow + 32, H);   // K = 32..63

    // ---------- pass A: softmax row sums ----------
    float rs[8];
#pragma unroll
    for (int r = 0; r < 8; ++r) rs[r] = 0.0f;

#pragma unroll 1
    for (int kt = 0; kt < NT; ++kt) {
        const _Float16* krow = kbase + (size_t)(kt * 16 + nlo) * Dd;
        v8f c = {};
        c = wmma_f16(a0, load_frag_f16(krow, H), c);
        c = wmma_f16(a1, load_frag_f16(krow + 32, H), c);
#pragma unroll
        for (int r = 0; r < 8; ++r)
            rs[r] += fast_exp2(__builtin_fmaf(c[r], S2, B2));
    }
    // fold normalization into the exponent: p = 2^(c*S2 + B2 - log2(rowsum))
    float brow[8];
#pragma unroll
    for (int r = 0; r < 8; ++r)
        brow[r] = B2 - __builtin_amdgcn_logf(half16_sum(rs[r]));

    // ---------- pass B: recompute, write score (NT), accumulate P*V ----------
    v8f o[4];
#pragma unroll
    for (int cI = 0; cI < 4; ++cI) o[cI] = (v8f){};

    _Float16* plds = &lds[wave * 640];   // 16 x 40 halves

#pragma unroll 1
    for (int kp = 0; kp < NT / 2; ++kp) {
#pragma unroll
        for (int half = 0; half < 2; ++half) {
            const int kt = kp * 2 + half;
            const _Float16* krow = kbase + (size_t)(kt * 16 + nlo) * Dd;
            v8f c = {};
            c = wmma_f16(a0, load_frag_f16(krow, H), c);
            c = wmma_f16(a1, load_frag_f16(krow + 32, H), c);
            float* srow = scb + kt * 16 + nlo;
#pragma unroll
            for (int r = 0; r < 8; ++r) {
                const int m = r + 8 * H;
                float p = fast_exp2(__builtin_fmaf(c[r], S2, brow[r]));
                __builtin_nontemporal_store(p, srow + (size_t)m * Nn);   // stream score
                plds[m * 40 + half * 16 + nlo] = (_Float16)p;            // stage for PV
            }
        }
        asm volatile("s_wait_dscnt 0x0" ::: "memory");

        // P tile (16x32) as A fragment: lane = query row, halves = key index
        v16h pa;
        {
            v8h lo = *(const v8h*)(plds + nlo * 40 + 8 * H);
            v8h hi = *(const v8h*)(plds + nlo * 40 + 16 + 8 * H);
#pragma unroll
            for (int j = 0; j < 8; ++j) { pa[j] = lo[j]; pa[j + 8] = hi[j]; }
        }
        const int key0 = kp * 32;
#pragma unroll
        for (int cI = 0; cI < 4; ++cI) {
            v16h vb = load_frag_f16(vbase + (size_t)(cI * 16 + nlo) * Nn + key0, H);
            o[cI] = wmma_f16(pa, vb, o[cI]);
        }
    }

    // ---------- write out ----------
#pragma unroll
    for (int cI = 0; cI < 4; ++cI)
#pragma unroll
        for (int r = 0; r < 8; ++r)
            ob[(size_t)(r + 8 * H) * Dd + cI * 16 + nlo] = o[cI][r];
}

extern "C" void kernel_launch(void* const* d_in, const int* in_sizes, int n_in,
                              void* d_out, int out_size, void* d_ws, size_t ws_size,
                              hipStream_t stream) {
    const float* q = (const float*)d_in[0];
    const float* k = (const float*)d_in[1];
    const float* v = (const float*)d_in[2];
    float* out   = (float*)d_out;
    float* score = out + (size_t)Bsz * Hn * Nn * Dd;      // tuple: (out, score)

    const size_t head_total = (size_t)Bsz * Hn * HEAD_ELEMS;   // 4 M elems
    _Float16* khp = (_Float16*)d_ws;                            // 8 MB
    _Float16* vtp = khp + head_total;                           // 8 MB

    kprep_kernel<<<dim3((unsigned)(head_total / 4 / 256)), dim3(256), 0, stream>>>(k, khp);
    vtrans_kernel<<<dim3(Bsz * Hn * (Nn / 64)), dim3(256), 0, stream>>>(v, vtp);
    hamm_attn_kernel<<<dim3(512), dim3(256), 0, stream>>>(q, khp, vtp, out, score);
}
